// ModulatedDeformConv3DWithOffset_60215441489982
// MI455X (gfx1250) — matrix operand order repaired and checked by reading, not compiled, not run on hardware
//
#include <hip/hip_runtime.h>
#include <hip/hip_bf16.h>

typedef __attribute__((ext_vector_type(16))) __bf16 v16bf;
typedef __attribute__((ext_vector_type(8)))  float  v8f;
typedef int v4i_g __attribute__((__vector_size__(4 * sizeof(int))));

struct Samp { float w0, w1, w2, w3; int i0, i1, i2, i3; };  // 32 bytes

union FragU { uint4 u4[2]; v16bf v; };

// ---- CDNA5 async global->LDS path (guarded: falls back to VGPR staging) ----
#if defined(__has_builtin)
#if __has_builtin(__builtin_amdgcn_global_load_async_to_lds_b128)
#define HAVE_ASYNC_LDS 1
#endif
#endif

// Builtin expects: (int4 AS1*, int4 AS3*, imm offset, imm cpol)
#define AS1_V4I(p) ((__attribute__((address_space(1))) v4i_g*)(uintptr_t)(p))
#define AS3_V4I(p) ((__attribute__((address_space(3))) v4i_g*)(uintptr_t)(p))

__device__ __forceinline__ void wait_async_zero() {
#if defined(__has_builtin)
#if __has_builtin(__builtin_amdgcn_s_wait_asynccnt)
    __builtin_amdgcn_s_wait_asynccnt(0);
    return;
#else
    asm volatile("s_wait_asynccnt 0x0" ::: "memory");
    return;
#endif
#else
    asm volatile("s_wait_asynccnt 0x0" ::: "memory");
#endif
}

// x logical layout (B=2, C=256, T=8, H=32, W=32); image n = b*8+t
// base(n,c) = ((b*256+c)*8+t)*1024  = (b*2048+t)*1024 + c*8192
__device__ __forceinline__ size_t img_base(int n) {
    int b = n >> 3, t = n & 7;
    return (size_t)(b * 2048 + t) * 1024;
}

// ---------------------------------------------------------------------------
// Kernel 1: offset conv (3x3, 256 -> 27) fused with offset post-processing.
// One thread per output pixel; produces 9 sample descriptors per pixel.
// ---------------------------------------------------------------------------
__global__ __launch_bounds__(256)
void k_offset_samp(const float* __restrict__ x, const float* __restrict__ ow,
                   const float* __restrict__ ob, Samp* __restrict__ samp) {
    int gid = blockIdx.x * 256 + threadIdx.x;   // 0..16383
    int n = gid >> 10, m = gid & 1023;
    int ho = m >> 5, wo = m & 31;
    size_t nbase = img_base(n);

    float acc[27];
#pragma unroll
    for (int j = 0; j < 27; ++j) acc[j] = ob[j];

    for (int c = 0; c < 256; ++c) {
        const float* xc = x + nbase + (size_t)c * 8192;
        float xv[9];
#pragma unroll
        for (int ti = 0; ti < 3; ++ti) {
            int yy = ho - 1 + ti;
#pragma unroll
            for (int tj = 0; tj < 3; ++tj) {
                int xx = wo - 1 + tj;
                bool v = (yy >= 0) & (yy < 32) & (xx >= 0) & (xx < 32);
                xv[ti * 3 + tj] = v ? xc[yy * 32 + xx] : 0.0f;
            }
        }
        // ow layout (27, 256, 3, 3): ow[j][c][tap]
        const float* wc = ow + (size_t)c * 9;
#pragma unroll
        for (int j = 0; j < 27; ++j) {
            const float* wj = wc + (size_t)j * 2304;
#pragma unroll
            for (int tp = 0; tp < 9; ++tp) acc[j] += xv[tp] * wj[tp];
        }
    }

    // Post-process: offset channel q = om channel q (q<18), mask = 18..26.
    Samp* sp = samp + (size_t)gid * 9;
#pragma unroll
    for (int kk = 0; kk < 9; ++kk) {
        int ki = kk / 3, kj = kk % 3;
        float dy = acc[2 * kk];
        float dx = acc[2 * kk + 1];
        float msk = 1.0f / (1.0f + __expf(-acc[18 + kk]));
        float py = dy + (float)(ho - 1 + ki);
        float px = dx + (float)(wo - 1 + kj);
        float y0f = floorf(py), x0f = floorf(px);
        float ty = py - y0f, tx = px - x0f;
        int y0 = (int)y0f, x0 = (int)x0f;
        int y1 = y0 + 1, x1 = x0 + 1;
        float vy0 = (y0 >= 0 && y0 < 32) ? 1.0f : 0.0f;
        float vy1 = (y1 >= 0 && y1 < 32) ? 1.0f : 0.0f;
        float vx0 = (x0 >= 0 && x0 < 32) ? 1.0f : 0.0f;
        float vx1 = (x1 >= 0 && x1 < 32) ? 1.0f : 0.0f;
        int cy0 = min(max(y0, 0), 31), cy1 = min(max(y1, 0), 31);
        int cx0 = min(max(x0, 0), 31), cx1 = min(max(x1, 0), 31);
        Samp s;
        s.w0 = (1.0f - ty) * (1.0f - tx) * msk * vy0 * vx0;
        s.w1 = (1.0f - ty) * tx          * msk * vy0 * vx1;
        s.w2 = ty * (1.0f - tx)          * msk * vy1 * vx0;
        s.w3 = ty * tx                   * msk * vy1 * vx1;
        s.i0 = cy0 * 32 + cx0;
        s.i1 = cy0 * 32 + cx1;
        s.i2 = cy1 * 32 + cx0;
        s.i3 = cy1 * 32 + cx1;
        sp[kk] = s;
    }
}

// ---------------------------------------------------------------------------
// Kernel 2: repack weight (256o, 256c, 1, 3, 3) fp32 -> wBt[kk][o][c] bf16
// ---------------------------------------------------------------------------
__global__ __launch_bounds__(256)
void k_repack(const float* __restrict__ w, __bf16* __restrict__ wBt) {
    int idx = blockIdx.x * 256 + threadIdx.x;   // < 589824
    int c = idx & 255;
    int o = (idx >> 8) & 255;
    int kk = idx >> 16;
    wBt[((size_t)kk * 256 + o) * 256 + c] = (__bf16)w[((size_t)o * 256 + c) * 9 + kk];
}

// ---------------------------------------------------------------------------
// Kernel 3: fused bilinear-gather + GEMM with bf16 WMMA (f32 accumulate).
// Workgroup: 256 threads = 8 waves; output tile 64(M) x 256(N).
// K order: kk*256 + c ; K-block = 32 channels of one tap. 72 K-blocks.
// Double-buffered LDS tiles; one barrier per K-block. B tile streamed with
// GLOBAL_LOAD_ASYNC_TO_LDS_B128 (ASYNCcnt) when available -> never blocks
// the fragment ds_loads feeding the WMMAs.
// ---------------------------------------------------------------------------
#define AT_STRIDE 40   // bf16 elems per LDS row (80B, 16B aligned, conflict-spread)

__device__ __forceinline__ void fill_B(const __bf16* __restrict__ wBt,
                                       __bf16* ldsRow, int kk, int cb, int tid) {
    const __bf16* gsrc = wBt + ((size_t)kk * 256 + tid) * 256 + cb;   // 64B
#if defined(HAVE_ASYNC_LDS)
    __builtin_amdgcn_global_load_async_to_lds_b128(AS1_V4I(gsrc), AS3_V4I(ldsRow), 0, 0);
    __builtin_amdgcn_global_load_async_to_lds_b128(AS1_V4I(gsrc), AS3_V4I(ldsRow), 16, 0);
    __builtin_amdgcn_global_load_async_to_lds_b128(AS1_V4I(gsrc), AS3_V4I(ldsRow), 32, 0);
    __builtin_amdgcn_global_load_async_to_lds_b128(AS1_V4I(gsrc), AS3_V4I(ldsRow), 48, 0);
#else
    const uint4* src = reinterpret_cast<const uint4*>(gsrc);
    uint4* dst = reinterpret_cast<uint4*>(ldsRow);
#pragma unroll
    for (int i = 0; i < 4; ++i) dst[i] = src[i];
#endif
}

__global__ __launch_bounds__(256)
void k_main(const float* __restrict__ x, const Samp* __restrict__ samp,
            const __bf16* __restrict__ wBt, const float* __restrict__ bias,
            float* __restrict__ out) {
    __shared__ __bf16 At[2][64 * AT_STRIDE];    // 10 KB
    __shared__ __bf16 Bt[2][256 * AT_STRIDE];   // 40 KB
    __shared__ Samp   Sm[576];                  // 18 KB

    int tid = threadIdx.x, lane = tid & 31, wv = tid >> 5;
    int tile = blockIdx.x;                   // 256 tiles = 16 n * 16 m-tiles
    int n = tile >> 4;
    int m0 = (tile & 15) << 6;
    size_t nbase = img_base(n);
    int b = n >> 3, t = n & 7;

    // Stage 576 sample descriptors for this M-tile.
    const Samp* sg = samp + ((size_t)n * 1024 + m0) * 9;
    for (int i = tid; i < 576; i += 256) Sm[i] = sg[i];
    __syncthreads();

    v8f acc[8] = {};
    int mSub  = wv >> 1;           // 0..3 : which 16-row M subtile
    int jBase = (wv & 1) * 8;      // 0 or 8 : which half of the 16 N-tiles
    int hl    = lane >> 4;         // lane half
    int aRow  = mSub * 16 + (lane & 15);

    // ---- prologue: fill buffer 0 with K-block 0 (kk=0, cb=0) ----
    {
        fill_B(wBt, &Bt[0][tid * AT_STRIDE], 0, 0, tid);
#pragma unroll
        for (int it = 0; it < 4; ++it) {
            int m = lane + ((it & 1) << 5);
            int p = wv + ((it >> 1) << 3);
            int c = 2 * p;
            Samp s = Sm[m * 9];
            const float* xb = x + nbase + (size_t)c * 8192;
            float v0 = s.w0 * xb[s.i0] + s.w1 * xb[s.i1] +
                       s.w2 * xb[s.i2] + s.w3 * xb[s.i3];
            const float* xb1 = xb + 8192;
            float v1 = s.w0 * xb1[s.i0] + s.w1 * xb1[s.i1] +
                       s.w2 * xb1[s.i2] + s.w3 * xb1[s.i3];
            At[0][m * AT_STRIDE + 2 * p]     = (__bf16)v0;
            At[0][m * AT_STRIDE + 2 * p + 1] = (__bf16)v1;
        }
        wait_async_zero();
    }
    __syncthreads();

    for (int blk = 0; blk < 72; ++blk) {
        int cur = blk & 1;
        int nxt = cur ^ 1;
        bool hasNext = (blk + 1) < 72;
        int kkN = (blk + 1) >> 3;            // next block's tap
        int cbN = ((blk + 1) & 7) << 5;      // next block's channel base

        // -- (a) next block's global work: B tile async->LDS, gather->regs --
        float g0[4], g1[4];
        if (hasNext) {
            fill_B(wBt, &Bt[nxt][tid * AT_STRIDE], kkN, cbN, tid);
#pragma unroll
            for (int it = 0; it < 4; ++it) {
                int m = lane + ((it & 1) << 5);
                int p = wv + ((it >> 1) << 3);
                int c = cbN + 2 * p;
                Samp s = Sm[m * 9 + kkN];
                const float* xb = x + nbase + (size_t)c * 8192;
                g0[it] = s.w0 * xb[s.i0] + s.w1 * xb[s.i1] +
                         s.w2 * xb[s.i2] + s.w3 * xb[s.i3];
                const float* xb1 = xb + 8192;
                g1[it] = s.w0 * xb1[s.i0] + s.w1 * xb1[s.i1] +
                         s.w2 * xb1[s.i2] + s.w3 * xb1[s.i3];
            }
            if (blk + 2 < 72) {   // prefetch weight tile two blocks ahead
                int kkP = (blk + 2) >> 3, cbP = ((blk + 2) & 7) << 5;
                __builtin_prefetch(wBt + ((size_t)kkP * 256 + tid) * 256 + cbP, 0, 1);
            }
        }

        // -- (b) ALL fragment ds_loads for current buffer (before any ds_store) --
        FragU af;
        {
            const uint4* pA = reinterpret_cast<const uint4*>(
                &At[cur][aRow * AT_STRIDE + hl * 16]);
            af.u4[0] = pA[0];       // K base .. +7
            af.u4[1] = pA[1];       // K base+8 .. +15
        }
        FragU bfr[8];
#pragma unroll
        for (int j = 0; j < 8; ++j) {
            int bRow = (jBase + j) * 16 + (lane & 15);
            const uint4* q = reinterpret_cast<const uint4*>(
                &Bt[cur][bRow * AT_STRIDE + hl * 16]);
            bfr[j].u4[0] = q[0];
            bfr[j].u4[1] = q[1];
        }

        // -- (c) ds_stores of next A tile (waits only on gather loads) --
        if (hasNext) {
#pragma unroll
            for (int it = 0; it < 4; ++it) {
                int m = lane + ((it & 1) << 5);
                int p = wv + ((it >> 1) << 3);
                At[nxt][m * AT_STRIDE + 2 * p]     = (__bf16)g0[it];
                At[nxt][m * AT_STRIDE + 2 * p + 1] = (__bf16)g1[it];
            }
        }

        // -- (d) 8 back-to-back WMMAs on registered fragments --
#pragma unroll
        for (int j = 0; j < 8; ++j) {
            acc[j] = __builtin_amdgcn_wmma_f32_16x16x32_bf16(
                false, af.v, false, bfr[j].v, (short)0, acc[j], false, false);
        }

        wait_async_zero();    // async B-tile writes visible before barrier
        __syncthreads();
    }

    // ---- Store: C/D layout lanes 0-15 -> M=r, lanes 16-31 -> M=8+r ----
    int col = lane & 15;
    int mrow = m0 + mSub * 16 + hl * 8;
#pragma unroll
    for (int j = 0; j < 8; ++j) {
        int o = (jBase + j) * 16 + col;
        float bs = bias[o];
        float* op = out + ((size_t)(b * 256 + o) * 8 + t) * 1024 + mrow;
        float4 lo = make_float4(acc[j][0] + bs, acc[j][1] + bs,
                                acc[j][2] + bs, acc[j][3] + bs);
        float4 hi = make_float4(acc[j][4] + bs, acc[j][5] + bs,
                                acc[j][6] + bs, acc[j][7] + bs);
        *reinterpret_cast<float4*>(op)     = lo;
        *reinterpret_cast<float4*>(op + 4) = hi;
    }
}

// ---------------------------------------------------------------------------
extern "C" void kernel_launch(void* const* d_in, const int* in_sizes, int n_in,
                              void* d_out, int out_size, void* d_ws, size_t ws_size,
                              hipStream_t stream) {
    (void)in_sizes; (void)n_in; (void)out_size; (void)ws_size;
    const float* x      = (const float*)d_in[0];
    const float* weight = (const float*)d_in[1];
    const float* bias   = (const float*)d_in[2];
    const float* ow     = (const float*)d_in[3];
    const float* ob     = (const float*)d_in[4];

    char* ws = (char*)d_ws;
    Samp*   samp = (Samp*)ws;                                   // 4,718,592 B
    __bf16* wBt  = (__bf16*)(ws + 16384ull * 9 * sizeof(Samp)); // 1,179,648 B

    float* out = (float*)d_out;

    k_offset_samp<<<64, 256, 0, stream>>>(x, ow, ob, samp);
    k_repack<<<2304, 256, 0, stream>>>(weight, wBt);
    k_main<<<256, 256, 0, stream>>>(x, samp, wBt, bias, out);
}
